// EntModel_2482491097290
// MI455X (gfx1250) — matrix operand order repaired and checked by reading, not compiled, not run on hardware
//
#include <hip/hip_runtime.h>
#include <hip/hip_bf16.h>

typedef __attribute__((ext_vector_type(16))) _Float16 v16h;
typedef __attribute__((ext_vector_type(8)))  _Float16 v8h;
typedef __attribute__((ext_vector_type(8)))  float    v8f;

#define Bv   64
#define Tv   512
#define Ev   300
#define Ep   320
#define HDv  256
#define Gv   1024   // 4*HD
#define Kv   20
#define Hv   512
#define EPAD 32     // padded emission cols

// ---------------- WMMA helpers (CDNA5 16x16x32 f16 -> f32) ----------------

static __device__ inline v8f wmma32(v16h a, v16h b, v8f c) {
    // 8 args: (neg_a, A, neg_b, B, c_mod, C, reuse_a, reuse_b)
    return __builtin_amdgcn_wmma_f32_16x16x32_f16(false, a, false, b, (short)0, c, false, false);
}

// A-fragment (16-bit A 16x32), contiguous K as used here: lane holds row (lane&15),
// halves 0..7 = K sel*16..sel*16+7, halves 8..15 = K sel*16+8..sel*16+15 (B loaded the same).
static __device__ inline v16h make_a(const _Float16* lo, const _Float16* hi) {
    v8h l = *(const v8h*)lo;
    v8h h = *(const v8h*)hi;
    v16h a;
#pragma unroll
    for (int i = 0; i < 8; ++i) { a[i] = l[i]; a[i + 8] = h[i]; }
    return a;
}

static __device__ inline float sigm(float x) { return 1.0f / (1.0f + expf(-x)); }

// ---------------- 1) fp32 -> f16 pad/convert for weights ----------------

__global__ void k_pad16(const float* __restrict__ src, _Float16* __restrict__ dst,
                        int rows, int cols, int prows, int pcols) {
    int i = blockIdx.x * blockDim.x + threadIdx.x;
    int tot = prows * pcols;
    if (i >= tot) return;
    int r = i / pcols, c = i % pcols;
    float v = (r < rows && c < cols) ? src[r * cols + c] : 0.0f;
    dst[i] = (_Float16)v;
}

// ---------------- 2) embedding gather -> f16, K-padded to 320 ----------------

__global__ void k_embed(const int* __restrict__ sent, const float* __restrict__ tab,
                        _Float16* __restrict__ embf) {
    int row = blockIdx.x;        // token row = b*T + t
    int e   = threadIdx.x;       // 0..319
    int id  = sent[row];
    float v = (e < Ev) ? tab[(long)id * Ev + e] : 0.0f;
    embf[(long)row * Ep + e] = (_Float16)v;
}

// ---------------- 3) input projection GEMM: Gx[BT,1024] = emb[BT,320] @ Wih16^T ----
// One 16x16 output tile per wave, K = 320 = 10 chunks of 32.

__global__ void k_xproj(const _Float16* __restrict__ embf,
                        const _Float16* __restrict__ W,     // [1024,320] f16
                        _Float16* __restrict__ Gx) {        // [BT,1024] f16
    int wave = threadIdx.x >> 5, lane = threadIdx.x & 31;
    int sel = lane >> 4, l15 = lane & 15;
    int tid = blockIdx.x * 8 + wave;          // 131072 tiles total
    int mt = tid >> 6, nt = tid & 63;
    int m0 = mt * 16, n0 = nt * 16;

    const _Float16* arow = embf + (long)(m0 + l15) * Ep;
    const _Float16* brow = W    + (long)(n0 + l15) * Ep;
    v8f acc = {};
#pragma unroll
    for (int kk = 0; kk < 10; ++kk) {
        int kc = kk * 32;
        v16h a = make_a(arow + kc + sel * 16, arow + kc + sel * 16 + 8);
        v16h b = *(const v16h*)(brow + kc + sel * 16);
        acc = wmma32(a, b, acc);
    }
#pragma unroll
    for (int r = 0; r < 8; ++r) {
        int row = m0 + r + 8 * sel;
        Gx[(long)row * Gv + n0 + l15] = (_Float16)acc[r];
    }
}

// ---------------- 4) LSTM recurrence ----------------
// grid = 8 blocks: dir = blockIdx.x/4, batch slice of 16 rows = blockIdx.x%4.
// 512 threads = 16 waves; wave w owns hidden tile w (cols w*16..w*16+15) of all
// four gates for its 16 batch rows. h double-buffered f16 in LDS (stride 264
// halves => 4-bank row step, conflict-free b128 A loads); c in registers.

__global__ __launch_bounds__(512)
void k_lstm(const _Float16* __restrict__ GxF, const _Float16* __restrict__ GxB,
            const _Float16* __restrict__ WhF, const _Float16* __restrict__ WhB,
            const float* __restrict__ bF, const float* __restrict__ bB,
            const int* __restrict__ lengths,
            _Float16* __restrict__ hF, _Float16* __restrict__ hB) {
    extern __shared__ _Float16 hsmem[];            // 2 * 16*264 halves
    __shared__ int llen[16];

    const int dir   = blockIdx.x >> 2;
    const int boff  = (blockIdx.x & 3) * 16;       // global batch offset
    const _Float16* Gx   = dir ? GxB : GxF;
    const _Float16* Wh   = dir ? WhB : WhF;        // [1024,256] f16
    const float*    bias = dir ? bB  : bF;
    _Float16*       hOut = dir ? hB  : hF;         // [B,T,HD] f16

    _Float16* hbuf0 = hsmem;
    _Float16* hbuf1 = hsmem + 16 * 264;

    for (int i = threadIdx.x; i < 2 * 16 * 264; i += 512) hsmem[i] = (_Float16)0.0f;
    if (threadIdx.x < 16) llen[threadIdx.x] = lengths[boff + threadIdx.x];
    __syncthreads();

    const int wave = threadIdx.x >> 5, lane = threadIdx.x & 31;
    const int sel = lane >> 4, l15 = lane & 15;
    const int hcol0 = wave * 16;
    const int j = hcol0 + l15;                     // hidden column of this lane

    float cst[8];
#pragma unroll
    for (int r = 0; r < 8; ++r) cst[r] = 0.0f;
    float bpre[4];
#pragma unroll
    for (int g = 0; g < 4; ++g) bpre[g] = bias[g * HDv + j];

    const _Float16* wbase = Wh + (long)(hcol0 + l15) * HDv;   // gate row stride = 256*256

    for (int s = 0; s < Tv; ++s) {
        const _Float16* hc = (s & 1) ? hbuf1 : hbuf0;
        _Float16*       hn = (s & 1) ? hbuf0 : hbuf1;

        v8f a0 = {}, a1 = {}, a2 = {}, a3 = {};
        const _Float16* hrow = hc + l15 * 264;
#pragma unroll
        for (int kk = 0; kk < 8; ++kk) {
            int kc = kk * 32;
            v16h a = make_a(hrow + kc + sel * 16, hrow + kc + sel * 16 + 8);
            const _Float16* wb = wbase + kc + sel * 16;
            a0 = wmma32(a, *(const v16h*)(wb + 0 * HDv * HDv), a0);
            a1 = wmma32(a, *(const v16h*)(wb + 1 * HDv * HDv), a1);
            a2 = wmma32(a, *(const v16h*)(wb + 2 * HDv * HDv), a2);
            a3 = wmma32(a, *(const v16h*)(wb + 3 * HDv * HDv), a3);
        }

#pragma unroll
        for (int r = 0; r < 8; ++r) {
            int bl = 8 * sel + r;                  // local batch row (C/D layout)
            int bg = boff + bl;                    // global batch row
            int len = llen[bl];
            int tok = dir ? ((s < len) ? (len - 1 - s) : s) : s;
            long grow = ((long)bg * Tv + tok) * (long)Gv;
            float gi = a0[r] + (float)Gx[grow + 0 * HDv + j] + bpre[0];
            float gf = a1[r] + (float)Gx[grow + 1 * HDv + j] + bpre[1];
            float gg = a2[r] + (float)Gx[grow + 2 * HDv + j] + bpre[2];
            float go = a3[r] + (float)Gx[grow + 3 * HDv + j] + bpre[3];
            float c = sigm(gf) * cst[r] + sigm(gi) * tanhf(gg);
            float h = sigm(go) * tanhf(c);
            cst[r] = c;
            _Float16 hh = (_Float16)h;
            hn[bl * 264 + j] = hh;                                  // next-step LDS h
            hOut[((long)bg * Tv + tok) * HDv + j] = hh;             // global h
        }
        __syncthreads();   // writes to hn visible before anyone reads it at s+1
    }
}

// ---------------- 5) emission GEMM: emis[BT,32] = [hF|hB] @ Wout16^T + b_out ----

__global__ void k_emis(const _Float16* __restrict__ hFm, const _Float16* __restrict__ hBm,
                       const _Float16* __restrict__ Wo,   // [32,512] f16, rows 20..31 zero
                       const float* __restrict__ bout, float* __restrict__ emis) {
    int wave = threadIdx.x >> 5, lane = threadIdx.x & 31;
    int sel = lane >> 4, l15 = lane & 15;
    int tid = blockIdx.x * 8 + wave;       // 4096 tiles (2048 M x 2 N)
    int mt = tid >> 1, nt = tid & 1;
    int m0 = mt * 16, n0 = nt * 16;

    const _Float16* arF = hFm + (long)(m0 + l15) * HDv;
    const _Float16* arB = hBm + (long)(m0 + l15) * HDv;
    const _Float16* brow = Wo + (long)(n0 + l15) * Hv;
    v8f acc = {};
#pragma unroll
    for (int kk = 0; kk < 16; ++kk) {
        int kc = kk * 32;
        const _Float16* ap = (kc < HDv) ? (arF + kc) : (arB + kc - HDv);
        v16h a = make_a(ap + sel * 16, ap + sel * 16 + 8);
        v16h b = *(const v16h*)(brow + kc + sel * 16);
        acc = wmma32(a, b, acc);
    }
    int col = n0 + l15;
    float bv = (col < Kv) ? bout[col] : 0.0f;
#pragma unroll
    for (int r = 0; r < 8; ++r) {
        int row = m0 + r + 8 * sel;
        emis[(long)row * EPAD + col] = acc[r] + bv;
    }
}

// ---------------- 6) Viterbi decode: one wave per batch row ----------------

__global__ void k_viterbi(const float* __restrict__ emis, const int* __restrict__ lengths,
                          const float* __restrict__ trans, const int* __restrict__ stop_id,
                          float* __restrict__ out) {
    int b = blockIdx.x;
    int k = threadIdx.x;                   // 32 lanes, 20 active tags
    __shared__ float dsh[32];
    __shared__ float term[32];
    __shared__ unsigned char bps[Tv * Kv];

    float delta = 0.0f;                    // init_vvars all zeros (faithful)
    int len = lengths[b];

    for (int t = 0; t < Tv; ++t) {
        dsh[k] = delta;
        __syncthreads();
        if (k < Kv) {
            if (t < len) {
                float best = -3.0e38f; int arg = 0;
#pragma unroll
                for (int p = 0; p < Kv; ++p) {
                    float v = dsh[p] + trans[k * Kv + p];
                    if (v > best) { best = v; arg = p; }
                }
                delta = best + emis[((long)b * Tv + t) * EPAD + k];
                bps[t * Kv + k] = (unsigned char)arg;
            } else {
                bps[t * Kv + k] = (unsigned char)k;   // identity pass-through
            }
        }
        __syncthreads();
    }

    int stop = *stop_id;
    term[k] = (k < Kv) ? (delta + trans[stop * Kv + k]) : -3.0e38f;
    __syncthreads();

    if (k == 0) {
        float best = term[0]; int arg = 0;
        for (int i = 1; i < Kv; ++i)
            if (term[i] > best) { best = term[i]; arg = i; }
        out[b] = best;                                   // scores[b]
        float* paths = out + Bv + (long)b * (Tv + 1);
        paths[Tv] = (float)arg;
        int tag = arg;
        for (int t = Tv - 1; t >= 0; --t) {
            int prev = bps[t * Kv + tag];
            paths[t] = (float)prev;
            tag = prev;
        }
    }
}

// ---------------- host side ----------------

extern "C" void kernel_launch(void* const* d_in, const int* in_sizes, int n_in,
                              void* d_out, int out_size, void* d_ws, size_t ws_size,
                              hipStream_t stream) {
    (void)in_sizes; (void)n_in; (void)out_size; (void)ws_size;

    const int*   sentence = (const int*)d_in[0];
    const int*   lengths  = (const int*)d_in[1];
    const int*   stop_id  = (const int*)d_in[3];
    const float* table    = (const float*)d_in[4];
    const float* Wf_ih    = (const float*)d_in[5];
    const float* Wf_hh    = (const float*)d_in[6];
    const float* bf       = (const float*)d_in[7];
    const float* Wb_ih    = (const float*)d_in[8];
    const float* Wb_hh    = (const float*)d_in[9];
    const float* bb       = (const float*)d_in[10];
    const float* W_out    = (const float*)d_in[11];
    const float* b_out    = (const float*)d_in[12];
    const float* trans    = (const float*)d_in[13];

    char* wp = (char*)d_ws;
    auto carve = [&](size_t bytes) -> void* {
        void* p = (void*)wp;
        wp += (bytes + 255) & ~(size_t)255;
        return p;
    };
    const size_t BT = (size_t)Bv * Tv;
    _Float16* embf    = (_Float16*)carve(BT * Ep * 2);
    _Float16* Wfih16  = (_Float16*)carve((size_t)Gv * Ep * 2);
    _Float16* Wbih16  = (_Float16*)carve((size_t)Gv * Ep * 2);
    _Float16* Wfhh16  = (_Float16*)carve((size_t)Gv * HDv * 2);
    _Float16* Wbhh16  = (_Float16*)carve((size_t)Gv * HDv * 2);
    _Float16* Wout16  = (_Float16*)carve((size_t)32 * Hv * 2);
    _Float16* GxF     = (_Float16*)carve(BT * Gv * 2);
    _Float16* GxB     = (_Float16*)carve(BT * Gv * 2);
    _Float16* hF      = (_Float16*)carve(BT * HDv * 2);
    _Float16* hB      = (_Float16*)carve(BT * HDv * 2);
    float*    emisP   = (float*)   carve(BT * EPAD * 4);

    // weight conversion / padding
    k_pad16<<<(Gv * Ep + 255) / 256, 256, 0, stream>>>(Wf_ih, Wfih16, Gv, Ev, Gv, Ep);
    k_pad16<<<(Gv * Ep + 255) / 256, 256, 0, stream>>>(Wb_ih, Wbih16, Gv, Ev, Gv, Ep);
    k_pad16<<<(Gv * HDv + 255) / 256, 256, 0, stream>>>(Wf_hh, Wfhh16, Gv, HDv, Gv, HDv);
    k_pad16<<<(Gv * HDv + 255) / 256, 256, 0, stream>>>(Wb_hh, Wbhh16, Gv, HDv, Gv, HDv);
    k_pad16<<<(32 * Hv + 255) / 256, 256, 0, stream>>>(W_out, Wout16, Kv, Hv, 32, Hv);

    // embedding gather (f16, padded K)
    k_embed<<<(int)BT, Ep, 0, stream>>>(sentence, table, embf);

    // input projections: 2048 M-tiles x 64 N-tiles = 131072 tiles, 8 waves/block
    k_xproj<<<16384, 256, 0, stream>>>(embf, Wfih16, GxF);
    k_xproj<<<16384, 256, 0, stream>>>(embf, Wbih16, GxB);

    // recurrence: 8 persistent workgroups (2 dirs x 4 batch slices)
    size_t lds = (size_t)2 * 16 * 264 * sizeof(_Float16);
    k_lstm<<<8, 512, lds, stream>>>(GxF, GxB, Wfhh16, Wbhh16, bf, bb, lengths, hF, hB);

    // emissions: 2048 x 2 tiles, 8 waves/block
    k_emis<<<512, 256, 0, stream>>>(hF, hB, Wout16, b_out, emisP);

    // viterbi: one wave per batch row
    k_viterbi<<<Bv, 32, 0, stream>>>(emisP, lengths, trans, stop_id, (float*)d_out);
}
